// SGAT8_3496103379553
// MI455X (gfx1250) — compile-verified, matrix operand order checked by reading, not run on hardware
//
#include <hip/hip_runtime.h>

// SuperGAT x8 forward for MI455X (gfx1250, wave32).
// Dense projections via V_WMMA_F32_16X16X32_F16 (f16 A/B, f32 accum),
// A-tiles loaded with b128, B-tiles staged once per block in LDS.
// Edge phase: 32-edges-per-wave batching -> coalesced index/scalar loads,
// shfl-broadcast inner loop with 128B coalesced row gathers + L2-resident
// f32 atomics (hW = 12.8MB, int32 edge list = 25.6MB, both fit in 192MB L2).

typedef __attribute__((ext_vector_type(16))) _Float16 v16h;
typedef __attribute__((ext_vector_type(8)))  float    v8f;

// ---------------- WMMA tile loaders (layouts per CDNA5 ISA) -----
// 16-bit A 16x32: K(j,lane) = j + (lane>=16 ? 8 : 0) + (j>=8 ? 8 : 0).
// Two contiguous 8-float runs per lane -> four b128 loads.
__device__ __forceinline__ v16h load_a16(const float* __restrict__ rowptr, int kb) {
  const float4* lo = (const float4*)(rowptr + kb);        // K = kb .. kb+7
  const float4* hi = (const float4*)(rowptr + kb + 16);   // K = kb+16 .. kb+23
  float4 l0 = lo[0], l1 = lo[1], h0 = hi[0], h1 = hi[1];
  v16h a;
  a[0]  = (_Float16)l0.x; a[1]  = (_Float16)l0.y; a[2]  = (_Float16)l0.z; a[3]  = (_Float16)l0.w;
  a[4]  = (_Float16)l1.x; a[5]  = (_Float16)l1.y; a[6]  = (_Float16)l1.z; a[7]  = (_Float16)l1.w;
  a[8]  = (_Float16)h0.x; a[9]  = (_Float16)h0.y; a[10] = (_Float16)h0.z; a[11] = (_Float16)h0.w;
  a[12] = (_Float16)h1.x; a[13] = (_Float16)h1.y; a[14] = (_Float16)h1.z; a[15] = (_Float16)h1.w;
  return a;
}
// B 32x16 (KxN from row-major W[K][ldb]): same K striping, lane picks column.
// Used only by wave 0 of each block to stage LDS.
__device__ __forceinline__ v16h load_b16(const float* __restrict__ W, int ldb,
                                         int col, int kb) {
  v16h b;
#pragma unroll
  for (int j = 0; j < 16; ++j) {
    int K = j + kb + ((j >= 8) ? 8 : 0);
    b[j] = (_Float16)W[K * ldb + col];
  }
  return b;
}

#define WMMA_F16(A, B, C) \
  __builtin_amdgcn_wmma_f32_16x16x32_f16(false, (A), false, (B), (short)0, (C), false, false)

// ---------------- index narrowing -----------------------------------------
__global__ __launch_bounds__(256) void cvt_idx_kernel(const long long* __restrict__ idx64,
                                                      int* __restrict__ idx32, int n) {
  int i = blockIdx.x * blockDim.x + threadIdx.x;
  if (i < n) idx32[i] = (int)idx64[i];
}

// ---------------- input projection: h = x @ W0 + b0  (K=128) --------------
__global__ __launch_bounds__(256) void proj_in_kernel(const float* __restrict__ x,
    const float* __restrict__ W0, const float* __restrict__ b0,
    float* __restrict__ hbuf, int N) {
  __shared__ v16h Bsh[8 * 32];                     // [kstep*2 + tile][lane]
  int lane = threadIdx.x & 31;
  if (threadIdx.x < 32) {                          // wave 0 stages all B tiles
    int half = lane >> 4, r = lane & 15, kb = half * 8;
#pragma unroll
    for (int ks = 0; ks < 4; ++ks) {
      Bsh[(ks * 2 + 0) * 32 + lane] = load_b16(W0 + ks * 32 * 32, 32, r, kb);
      Bsh[(ks * 2 + 1) * 32 + lane] = load_b16(W0 + ks * 32 * 32, 32, 16 + r, kb);
    }
  }
  __syncthreads();
  int wid = (blockIdx.x * blockDim.x + threadIdx.x) >> 5;
  int m0 = wid * 16;
  if (m0 >= N) return;                             // N % 16 == 0: whole-wave exit
  int half = lane >> 4, r = lane & 15, kb = half * 8;
  const float* rowptr = x + (size_t)(m0 + r) * 128;
  v8f c0 = {}, c1 = {};
#pragma unroll
  for (int ks = 0; ks < 4; ++ks) {
    v16h a = load_a16(rowptr + ks * 32, kb);
    c0 = WMMA_F16(a, Bsh[(ks * 2 + 0) * 32 + lane], c0);
    c1 = WMMA_F16(a, Bsh[(ks * 2 + 1) * 32 + lane], c1);
  }
  int mrow = m0 + half * 8;
  float bias0 = b0[r], bias1 = b0[16 + r];
#pragma unroll
  for (int rr = 0; rr < 8; ++rr) {
    float* orow = hbuf + (size_t)(mrow + rr) * 32;
    orow[r]      = c0[rr] + bias0;
    orow[16 + r] = c1[rr] + bias1;
  }
}

// ---------------- per-layer projection: hW = h @ W  (K=32, no bias) -------
__global__ __launch_bounds__(256) void gemm32_kernel(const float* __restrict__ h,
    const float* __restrict__ W, float* __restrict__ hW, int N) {
  __shared__ v16h Bsh[2 * 32];
  int lane = threadIdx.x & 31;
  if (threadIdx.x < 32) {
    int half = lane >> 4, r = lane & 15, kb = half * 8;
    Bsh[lane]      = load_b16(W, 32, r, kb);
    Bsh[32 + lane] = load_b16(W, 32, 16 + r, kb);
  }
  __syncthreads();
  int wid = (blockIdx.x * blockDim.x + threadIdx.x) >> 5;
  int m0 = wid * 16;
  if (m0 >= N) return;
  int half = lane >> 4, r = lane & 15, kb = half * 8;
  v16h a = load_a16(h + (size_t)(m0 + r) * 32, kb);
  v8f c0 = {}, c1 = {};
  c0 = WMMA_F16(a, Bsh[lane], c0);
  c1 = WMMA_F16(a, Bsh[32 + lane], c1);
  int mrow = m0 + half * 8;
#pragma unroll
  for (int rr = 0; rr < 8; ++rr) {
    float* orow = hW + (size_t)(mrow + rr) * 32;
    orow[r]      = c0[rr];
    orow[16 + r] = c1[rr];
  }
}

// ---------------- output head: out = h @ W8 + b8  (32 -> 64) --------------
__global__ __launch_bounds__(256) void final_kernel(const float* __restrict__ h,
    const float* __restrict__ W8, const float* __restrict__ b8,
    float* __restrict__ out, int N) {
  __shared__ v16h Bsh[4 * 32];
  int lane = threadIdx.x & 31;
  if (threadIdx.x < 32) {
    int half = lane >> 4, r = lane & 15, kb = half * 8;
#pragma unroll
    for (int t = 0; t < 4; ++t)
      Bsh[t * 32 + lane] = load_b16(W8, 64, t * 16 + r, kb);
  }
  __syncthreads();
  int wid = (blockIdx.x * blockDim.x + threadIdx.x) >> 5;
  int m0 = wid * 16;
  if (m0 >= N) return;
  int half = lane >> 4, r = lane & 15, kb = half * 8;
  v16h a = load_a16(h + (size_t)(m0 + r) * 32, kb);
  v8f c[4];
#pragma unroll
  for (int t = 0; t < 4; ++t) {
    v8f cz = {};
    c[t] = WMMA_F16(a, Bsh[t * 32 + lane], cz);
  }
  int mrow = m0 + half * 8;
#pragma unroll
  for (int t = 0; t < 4; ++t) {
    float bias = b8[t * 16 + r];
#pragma unroll
    for (int rr = 0; rr < 8; ++rr)
      out[(size_t)(mrow + rr) * 64 + t * 16 + r] = c[t][rr] + bias;
  }
}

// ---------------- per-layer init ------------------------------------------
__global__ __launch_bounds__(256) void layer_init_kernel(float* __restrict__ agg,
    float* __restrict__ amax, float* __restrict__ denom, int N) {
  int i = blockIdx.x * blockDim.x + threadIdx.x;
  if (i < N * 32) agg[i] = 0.0f;
  if (i < N) { amax[i] = -3.402823466e38f; denom[i] = 0.0f; }
}

// ---------------- per-node attention scalars ------------------------------
__global__ __launch_bounds__(256) void node_att_kernel(const float* __restrict__ hW,
    const float* __restrict__ al, const float* __restrict__ ar,
    float* __restrict__ aL, float* __restrict__ aR, int N) {
  int n = blockIdx.x * blockDim.x + threadIdx.x;
  if (n >= N) return;
  const float4* row = (const float4*)(hW + (size_t)n * 32);
  const float4* l4 = (const float4*)al;
  const float4* r4 = (const float4*)ar;
  float sl = 0.0f, sr = 0.0f;
#pragma unroll
  for (int q = 0; q < 8; ++q) {
    float4 v = row[q], lv = l4[q], rv = r4[q];
    sl += v.x * lv.x + v.y * lv.y + v.z * lv.z + v.w * lv.w;
    sr += v.x * rv.x + v.y * rv.y + v.z * rv.z + v.w * rv.w;
  }
  aL[n] = sl;
  aR[n] = sr;
}

// ---------------- float atomic max (sign-split punning) -------------------
__device__ __forceinline__ void atomicMaxF(float* addr, float val) {
  if (val >= 0.0f) atomicMax((int*)addr, __float_as_int(val));
  else             atomicMin((unsigned int*)addr, __float_as_uint(val));
}

// ---------------- pass 1: alpha per edge + segment max --------------------
// 32 edges per wave batch: lane e loads its own edge scalars (coalesced),
// inner loop shfl-broadcasts s/d while all 32 lanes do the 128B row gather
// and the xor-tree dot-product reduction.
__global__ __launch_bounds__(256) void edge_alpha_kernel(const float* __restrict__ hW,
    const int* __restrict__ src, const int* __restrict__ dst,
    const float* __restrict__ aL, const float* __restrict__ aR,
    float* __restrict__ alpha, float* __restrict__ amax, int E) {
  int lane = threadIdx.x & 31;
  int wid  = (blockIdx.x * blockDim.x + threadIdx.x) >> 5;
  int nw   = (gridDim.x * blockDim.x) >> 5;
  for (int e0 = wid * 32; e0 < E; e0 += nw * 32) {
    int  myE   = e0 + lane;
    bool valid = myE < E;
    int  s_l = valid ? src[myE] : 0;
    int  d_l = valid ? dst[myE] : 0;
    float att_l = valid ? (aL[s_l] + aR[d_l]) : 0.0f;   // hj@att_l + hi@att_r
    float v_l = 0.0f;
    int cnt = E - e0; if (cnt > 32) cnt = 32;
    for (int i = 0; i < cnt; ++i) {
      int s = __shfl(s_l, i, 32);
      int d = __shfl(d_l, i, 32);
      float prod = hW[(size_t)s * 32 + lane] * hW[(size_t)d * 32 + lane];
#pragma unroll
      for (int off = 16; off > 0; off >>= 1)
        prod += __shfl_xor(prod, off, 32);              // all lanes get hi.hj
      float v = att_l * (1.0f / (1.0f + __expf(-prod)));
      v = (v > 0.0f) ? v : 0.2f * v;                    // leaky_relu
      v_l = (lane == i) ? v : v_l;                      // keep own edge's value
    }
    if (valid) {
      alpha[myE] = v_l;                                 // coalesced store
      atomicMaxF(amax + d_l, v_l);                      // one atomic per edge
    }
  }
}

// ---------------- pass 2: exp + segment sum -------------------------------
__global__ __launch_bounds__(256) void edge_exp_kernel(float* __restrict__ alpha,
    const int* __restrict__ dst, const float* __restrict__ amax,
    float* __restrict__ denom, int E) {
  int e = blockIdx.x * blockDim.x + threadIdx.x;
  if (e >= E) return;
  int d = dst[e];
  float ev = __expf(alpha[e] - amax[d]);
  alpha[e] = ev;
  atomicAdd(denom + d, ev);
}

// ---------------- pass 3: weighted scatter-sum ----------------------------
// Same 32-edge batching: per-lane coalesced scalar loads, then broadcast and
// do one coalesced 128B gather + 128B atomic-add burst per edge.
__global__ __launch_bounds__(256) void edge_aggr_kernel(const float* __restrict__ hW,
    const int* __restrict__ src, const int* __restrict__ dst,
    const float* __restrict__ alpha, const float* __restrict__ denom,
    float* __restrict__ agg, int E) {
  int lane = threadIdx.x & 31;
  int wid  = (blockIdx.x * blockDim.x + threadIdx.x) >> 5;
  int nw   = (gridDim.x * blockDim.x) >> 5;
  for (int e0 = wid * 32; e0 < E; e0 += nw * 32) {
    int  myE   = e0 + lane;
    bool valid = myE < E;
    int  s_l = valid ? src[myE] : 0;
    int  d_l = valid ? dst[myE] : 0;
    float a_l = valid ? alpha[myE] / (denom[d_l] + 1e-16f) : 0.0f;
    int cnt = E - e0; if (cnt > 32) cnt = 32;
    for (int i = 0; i < cnt; ++i) {
      int   s = __shfl(s_l, i, 32);
      int   d = __shfl(d_l, i, 32);
      float a = __shfl(a_l, i, 32);
      atomicAdd(agg + (size_t)d * 32 + lane, hW[(size_t)s * 32 + lane] * a);
    }
  }
}

// ---------------- epilogue: h = relu(agg + b) -----------------------------
__global__ __launch_bounds__(256) void node_finish_kernel(const float* __restrict__ agg,
    const float* __restrict__ b, float* __restrict__ hbuf, int N) {
  int i = blockIdx.x * blockDim.x + threadIdx.x;
  if (i >= N * 32) return;
  float v = agg[i] + b[i & 31];
  hbuf[i] = (v > 0.0f) ? v : 0.0f;
}

// ---------------- host orchestration --------------------------------------
extern "C" void kernel_launch(void* const* d_in, const int* in_sizes, int n_in,
                              void* d_out, int out_size, void* d_ws, size_t ws_size,
                              hipStream_t stream) {
  (void)n_in; (void)out_size; (void)ws_size;
  const float*     x    = (const float*)d_in[0];
  const long long* eidx = (const long long*)d_in[1];   // int64 [2,E]
  const float*     W0   = (const float*)d_in[2];
  const float*     b0   = (const float*)d_in[3];
  const float*     Wl   = (const float*)d_in[4];       // [8,32,32]
  const float*     bl   = (const float*)d_in[5];       // [8,32]
  const float*     attl = (const float*)d_in[6];       // [8,32]
  const float*     attr = (const float*)d_in[7];       // [8,32]
  const float*     W8   = (const float*)d_in[8];
  const float*     b8   = (const float*)d_in[9];

  const int N = in_sizes[0] / 128;
  const int E = in_sizes[1] / 2;

  // workspace carve-out (~78 MB): node feature buffers + edge scalars + int32 indices
  char* p = (char*)d_ws;
  float* hbuf  = (float*)p; p += (size_t)N * 32 * 4;
  float* hW    = (float*)p; p += (size_t)N * 32 * 4;
  float* agg   = (float*)p; p += (size_t)N * 32 * 4;
  float* alpha = (float*)p; p += (size_t)E * 4;
  float* aL    = (float*)p; p += (size_t)N * 4;
  float* aR    = (float*)p; p += (size_t)N * 4;
  float* amax  = (float*)p; p += (size_t)N * 4;
  float* denom = (float*)p; p += (size_t)N * 4;
  int*   idx32 = (int*)p;   p += (size_t)2 * E * 4;
  const int* src32 = idx32;
  const int* dst32 = idx32 + E;

  cvt_idx_kernel<<<(2 * E + 255) / 256, 256, 0, stream>>>(eidx, idx32, 2 * E);

  const int gblocks = ((N / 16) * 32 + 255) / 256;   // one wave per 16-row tile
  const int eblocks = 2048;                           // 32-edge batches per wave

  proj_in_kernel<<<gblocks, 256, 0, stream>>>(x, W0, b0, hbuf, N);

  for (int l = 0; l < 8; ++l) {
    const float* W  = Wl + (size_t)l * 32 * 32;
    const float* b  = bl + (size_t)l * 32;
    const float* al = attl + (size_t)l * 32;
    const float* ar = attr + (size_t)l * 32;

    layer_init_kernel<<<(N * 32 + 255) / 256, 256, 0, stream>>>(agg, amax, denom, N);
    gemm32_kernel<<<gblocks, 256, 0, stream>>>(hbuf, W, hW, N);
    node_att_kernel<<<(N + 255) / 256, 256, 0, stream>>>(hW, al, ar, aL, aR, N);
    edge_alpha_kernel<<<eblocks, 256, 0, stream>>>(hW, src32, dst32, aL, aR, alpha, amax, E);
    edge_exp_kernel<<<(E + 255) / 256, 256, 0, stream>>>(alpha, dst32, amax, denom, E);
    edge_aggr_kernel<<<eblocks, 256, 0, stream>>>(hW, src32, dst32, alpha, denom, agg, E);
    node_finish_kernel<<<(N * 32 + 255) / 256, 256, 0, stream>>>(agg, b, hbuf, N);
  }

  final_kernel<<<gblocks, 256, 0, stream>>>(hbuf, W8, b8, (float*)d_out, N);
}
